// SVD_Pi3Attention_48155173323025
// MI455X (gfx1250) — matrix operand order repaired and checked by reading, not compiled, not run on hardware
//
#include <hip/hip_runtime.h>

// ---------------------------------------------------------------------------
// SVD_Pi3Attention for MI455X (gfx1250, wave32, WMMA)
//   x:[4,2048,1024] f32, low-rank qkv (r=256), 16 heads x 64, low-rank out.
//   bf16 WMMA 16x16x32, f32 accumulation, flash attention with:
//     - async global->LDS staging for K blocks
//     - transposed-in-LDS V blocks (vector ds loads for B-fragments)
//     - softmax row-sum folded into the matrix engine via an all-ones tile
// ---------------------------------------------------------------------------

typedef __attribute__((ext_vector_type(16))) __bf16 v16bf;
typedef __attribute__((ext_vector_type(8)))  __bf16 v8bf;
typedef __attribute__((ext_vector_type(8)))  float  v8f;

#define BQ   4
#define TSEQ 2048
#define CDIM 1024
#define NH   16
#define HDIM 64
#define RQKV 256
#define C3   (3*CDIM)

__device__ __forceinline__ __bf16 f2bf(float f) {
    union { float f; unsigned u; } a; a.f = f;
    unsigned r = a.u + 0x7FFFu + ((a.u >> 16) & 1u);   // round-to-nearest-even
    union { unsigned short s; __bf16 b; } o; o.s = (unsigned short)(r >> 16);
    return o.b;
}

// Build a 16-element bf16 fragment from two 16-byte chunks (global or LDS).
__device__ __forceinline__ v16bf make_frag(const __bf16* p0, const __bf16* p1) {
    v8bf lo = *(const v8bf*)p0;
    v8bf hi = *(const v8bf*)p1;
    return __builtin_shufflevector(lo, hi, 0,1,2,3,4,5,6,7,8,9,10,11,12,13,14,15);
}

// ---------------------------------------------------------------------------
// f32 -> bf16 conversion (grid-stride)
// ---------------------------------------------------------------------------
__global__ void cvt_f32_bf16(const float* __restrict__ in,
                             __bf16* __restrict__ out, int n) {
    int i = blockIdx.x * blockDim.x + threadIdx.x;
    int stride = gridDim.x * blockDim.x;
    for (; i < n; i += stride) out[i] = f2bf(in[i]);
}

// ---------------------------------------------------------------------------
// C[M,N] = A[M,K] * B[N,K]^T   (both row-major, K contiguous)
// 256 threads = 8 waves; block tile 128x64; wave tile 32x32; K step 32.
// ---------------------------------------------------------------------------
template <bool F32OUT>
__global__ __launch_bounds__(256) void gemm_bf16_wmma(
    const __bf16* __restrict__ A, const __bf16* __restrict__ B,
    void* __restrict__ Cout, int M, int N, int K)
{
    const int lane = threadIdx.x & 31;
    const int wave = threadIdx.x >> 5;
    const int wm   = wave & 3;            // 4 waves along M
    const int wn   = wave >> 2;           // 2 waves along N
    const int m0   = blockIdx.y * 128 + wm * 32;
    const int n0   = blockIdx.x * 64  + wn * 32;
    const int lr   = lane & 15;
    const int lh   = lane >> 4;
    const int akb  = lh * 8;
    const int bkb  = lh * 8;

    v8f c[2][2] = {};
    for (int k0 = 0; k0 < K; k0 += 32) {
        v16bf a[2], b[2];
        #pragma unroll
        for (int i = 0; i < 2; ++i) {
            const __bf16* pa = A + (size_t)(m0 + i*16 + lr) * K + k0 + akb;
            a[i] = make_frag(pa, pa + 16);
            const __bf16* pb = B + (size_t)(n0 + i*16 + lr) * K + k0 + bkb;
            b[i] = make_frag(pb, pb + 16);
        }
        #pragma unroll
        for (int i = 0; i < 2; ++i)
            #pragma unroll
            for (int j = 0; j < 2; ++j)
                c[i][j] = __builtin_amdgcn_wmma_f32_16x16x32_bf16(
                    false, a[i], false, b[j], (short)0, c[i][j], false, false);
    }

    #pragma unroll
    for (int i = 0; i < 2; ++i)
        #pragma unroll
        for (int j = 0; j < 2; ++j) {
            size_t idx = (size_t)(m0 + i*16 + lh*8) * N + (n0 + j*16 + lr);
            if (F32OUT) {
                float* C = (float*)Cout;
                #pragma unroll
                for (int r = 0; r < 8; ++r) { C[idx] = c[i][j][r]; idx += N; }
            } else {
                __bf16* C = (__bf16*)Cout;
                #pragma unroll
                for (int r = 0; r < 8; ++r) { C[idx] = f2bf(c[i][j][r]); idx += N; }
            }
        }
}

// ---------------------------------------------------------------------------
// Flash attention over bf16 qkv [B, T, 3C].
// Grid: (T/64, B*H), 128 threads = 4 waves; each wave owns 16 query rows.
// ---------------------------------------------------------------------------
__global__ __launch_bounds__(128) void flash_attn_wmma(
    const __bf16* __restrict__ qkv, __bf16* __restrict__ y)
{
    __shared__ __bf16 lds_k[32][72];        // 32 keys x 64 dims (async target)
    __shared__ __bf16 lds_vt[64][40];       // V transposed: [d][key], padded
    __shared__ __bf16 lds_p[4][16][32];     // per-wave P tile (16 rows x 32 keys)

    const int tid  = threadIdx.x;
    const int lane = tid & 31;
    const int wave = tid >> 5;
    const int lr   = lane & 15;
    const int lh   = lane >> 4;

    const int bh = blockIdx.y;
    const int b  = bh >> 4;
    const int h  = bh & 15;
    const int q0 = blockIdx.x * 64 + wave * 16;

    const size_t rs = (size_t)C3;                         // row stride in elems
    const __bf16* base = qkv + (size_t)b * TSEQ * rs;
    const __bf16* qp = base + (size_t)h * HDIM;
    const __bf16* kp = base + CDIM  + (size_t)h * HDIM;
    const __bf16* vp = base + 2*CDIM + (size_t)h * HDIM;

    // Q fragments: D=64 -> two 16x32 A-frags, loaded once.
    v16bf qa[2];
    #pragma unroll
    for (int s = 0; s < 2; ++s) {
        const __bf16* p = qp + (size_t)(q0 + lr) * rs + s*32 + lh*8;
        qa[s] = make_frag(p, p + 16);
    }

    // All-ones B tile: folds the softmax row-sum into the PV matmul
    // (every column of o[4] becomes the running row sum l, pre-scaled by
    //  alpha each step exactly like the other O accumulators).
    v16bf ones;
    #pragma unroll
    for (int j = 0; j < 16; ++j) ones[j] = f2bf(1.0f);

    v8f   o[5] = {};                        // o[0..3]: output d-tiles, o[4]: l
    float m[8];
    #pragma unroll
    for (int j = 0; j < 8; ++j) m[j] = -3.0e38f;

    const float scale = 0.125f;             // 1/sqrt(64)
    const int vrow = tid >> 2;              // 0..31 (staging key row)
    const int vseg = (tid & 3) * 16;        // 0/16/32/48 (staging d segment)

    // LDS byte offset for the async K copy target (generic LDS pointer bits
    // [31:0] are the wave-relative LDS offset per the CDNA5 aperture rules).
    const unsigned lkoff = (unsigned)(size_t)(void*)&lds_k[vrow][vseg];

    for (int kb = 0; kb < TSEQ; kb += 32) {
        __syncthreads();                    // previous block's LDS reads done
        // ---- K: async global->LDS (32B/thread); V: load + transpose ----
        {
            const int goff = (int)(((kb + vrow) * (int)rs + vseg) * 2); // bytes
            asm volatile(
                "global_load_async_to_lds_b128 %0, %1, %2 offset:0\n\t"
                "global_load_async_to_lds_b128 %0, %1, %2 offset:16"
                :
                : "v"(lkoff), "v"(goff), "s"(kp)
                : "memory");
            const __bf16* gv = vp + (size_t)(kb + vrow) * rs + vseg;
            v8bf vlo = *(const v8bf*)gv;
            v8bf vhi = *(const v8bf*)(gv + 8);
            #pragma unroll
            for (int i = 0; i < 8; ++i) {
                lds_vt[vseg + i][vrow]     = vlo[i];
                lds_vt[vseg + 8 + i][vrow] = vhi[i];
            }
            if (kb + 32 < TSEQ) {           // prefetch next block toward L2
                __builtin_prefetch(kp + (size_t)(kb + 32 + vrow) * rs + vseg, 0, 1);
                __builtin_prefetch(vp + (size_t)(kb + 32 + vrow) * rs + vseg, 0, 1);
            }
            asm volatile("s_wait_asynccnt 0" ::: "memory");
        }
        __syncthreads();

        // ---- S = Q * K^T : load all 4 K-frags first, then 4 WMMAs ----
        v16bf bk[2][2];
        #pragma unroll
        for (int nt = 0; nt < 2; ++nt)
            #pragma unroll
            for (int s = 0; s < 2; ++s) {
                const __bf16* pk = &lds_k[nt*16 + lr][s*32 + lh*8];
                bk[nt][s] = make_frag(pk, pk + 16);
            }
        v8f s0 = {}, s1 = {};
        s0 = __builtin_amdgcn_wmma_f32_16x16x32_bf16(false, qa[0], false, bk[0][0], (short)0, s0, false, false);
        s1 = __builtin_amdgcn_wmma_f32_16x16x32_bf16(false, qa[0], false, bk[1][0], (short)0, s1, false, false);
        s0 = __builtin_amdgcn_wmma_f32_16x16x32_bf16(false, qa[1], false, bk[0][1], (short)0, s0, false, false);
        s1 = __builtin_amdgcn_wmma_f32_16x16x32_bf16(false, qa[1], false, bk[1][1], (short)0, s1, false, false);

        // ---- online softmax: only the row-max needs cross-lane reduction ----
        float p0[8], p1[8], alpha[8];
        #pragma unroll
        for (int j = 0; j < 8; ++j) {
            float a  = s0[j] * scale;
            float bb = s1[j] * scale;
            float t  = fmaxf(a, bb);
            #pragma unroll
            for (int off = 1; off < 16; off <<= 1)
                t = fmaxf(t, __shfl_xor(t, off, 16));
            float mn = fmaxf(m[j], t);
            alpha[j] = __expf(m[j] - mn);
            m[j] = mn;
            p0[j] = __expf(a - mn);
            p1[j] = __expf(bb - mn);
        }
        #pragma unroll
        for (int t5 = 0; t5 < 5; ++t5)
            #pragma unroll
            for (int j = 0; j < 8; ++j) o[t5][j] *= alpha[j];

        // ---- P tile -> LDS (C-layout to row-major); per-wave slice, and
        //      same-wave LDS ops are in-order, so no barrier needed here ----
        #pragma unroll
        for (int j = 0; j < 8; ++j) {
            lds_p[wave][lh*8 + j][lr]      = f2bf(p0[j]);
            lds_p[wave][lh*8 + j][16 + lr] = f2bf(p1[j]);
        }
        const __bf16* pp = &lds_p[wave][lr][lh*8];
        v16bf pA = make_frag(pp, pp + 16);

        // ---- O += P * V : vector B-frags from transposed V, + ones tile ----
        #pragma unroll
        for (int t4 = 0; t4 < 4; ++t4) {
            const __bf16* pv = &lds_vt[t4*16 + lr][lh*8];
            v16bf vb = make_frag(pv, pv + 16);
            o[t4] = __builtin_amdgcn_wmma_f32_16x16x32_bf16(
                false, pA, false, vb, (short)0, o[t4], false, false);
        }
        o[4] = __builtin_amdgcn_wmma_f32_16x16x32_bf16(
            false, pA, false, ones, (short)0, o[4], false, false);
    }

    // ---- normalize (l lives in o[4], identical across columns) and store ----
    float inv[8];
    #pragma unroll
    for (int j = 0; j < 8; ++j) inv[j] = 1.0f / o[4][j];
    #pragma unroll
    for (int t4 = 0; t4 < 4; ++t4)
        #pragma unroll
        for (int j = 0; j < 8; ++j) {
            int row = q0 + lh*8 + j;
            int d   = t4*16 + lr;
            y[(size_t)(b*TSEQ + row) * CDIM + h*HDIM + d] = f2bf(o[t4][j] * inv[j]);
        }
}

// ---------------------------------------------------------------------------
extern "C" void kernel_launch(void* const* d_in, const int* in_sizes, int n_in,
                              void* d_out, int out_size, void* d_ws, size_t ws_size,
                              hipStream_t stream) {
    (void)in_sizes; (void)n_in; (void)out_size; (void)ws_size;
    const float* x   = (const float*)d_in[0];   // [B,T,C]
    const float* Wv  = (const float*)d_in[1];   // [R, C]
    const float* Wu  = (const float*)d_in[2];   // [3C, R]
    const float* Wov = (const float*)d_in[3];   // [R, C]
    const float* Wou = (const float*)d_in[4];   // [C, R]

    const int M = BQ * TSEQ;                    // 8192

    char* cur = (char*)d_ws;
    auto carve = [&](size_t bytes) -> char* {
        char* p = cur; cur += (bytes + 255) & ~(size_t)255; return p;
    };
    __bf16* xb   = (__bf16*)carve((size_t)M * CDIM * 2);
    __bf16* wvb  = (__bf16*)carve((size_t)RQKV * CDIM * 2);
    __bf16* wub  = (__bf16*)carve((size_t)C3 * RQKV * 2);
    __bf16* wovb = (__bf16*)carve((size_t)RQKV * CDIM * 2);
    __bf16* woub = (__bf16*)carve((size_t)CDIM * RQKV * 2);
    __bf16* t    = (__bf16*)carve((size_t)M * RQKV * 2);
    __bf16* qkvb = (__bf16*)carve((size_t)M * C3 * 2);
    __bf16* yb   = (__bf16*)carve((size_t)M * CDIM * 2);
    __bf16* y1   = (__bf16*)carve((size_t)M * RQKV * 2);

    // 1. convert inputs to bf16
    cvt_f32_bf16<<<2048, 256, 0, stream>>>(x,   xb,   M * CDIM);
    cvt_f32_bf16<<<512,  256, 0, stream>>>(Wv,  wvb,  RQKV * CDIM);
    cvt_f32_bf16<<<512,  256, 0, stream>>>(Wu,  wub,  C3 * RQKV);
    cvt_f32_bf16<<<512,  256, 0, stream>>>(Wov, wovb, RQKV * CDIM);
    cvt_f32_bf16<<<512,  256, 0, stream>>>(Wou, woub, CDIM * RQKV);

    // 2. t = x * Wv^T            [8192,1024] x [256,1024]^T
    gemm_bf16_wmma<false><<<dim3(RQKV/64, M/128), 256, 0, stream>>>(
        xb, wvb, t, M, RQKV, CDIM);
    // 3. qkv = t * Wu^T          [8192,256] x [3072,256]^T
    gemm_bf16_wmma<false><<<dim3(C3/64, M/128), 256, 0, stream>>>(
        t, wub, qkvb, M, C3, RQKV);
    // 4. flash attention -> y    [8192,1024]
    flash_attn_wmma<<<dim3(TSEQ/64, BQ*NH), 128, 0, stream>>>(qkvb, yb);
    // 5. y1 = y * Wov^T          [8192,1024] x [256,1024]^T
    gemm_bf16_wmma<false><<<dim3(RQKV/64, M/128), 256, 0, stream>>>(
        yb, wovb, y1, M, RQKV, CDIM);
    // 6. out = y1 * Wou^T -> f32 [8192,256] x [1024,256]^T
    gemm_bf16_wmma<true><<<dim3(CDIM/64, M/128), 256, 0, stream>>>(
        y1, woub, (float*)d_out, M, CDIM, RQKV);
}
